// MemoryGraph_32719060861003
// MI455X (gfx1250) — matrix operand order repaired and checked by reading, not compiled, not run on hardware
//
#include <hip/hip_runtime.h>
#include <hip/hip_bf16.h>

// Problem constants (from reference)
#define BB    8
#define NC    64
#define NN    128
#define DN    64
#define HS    256
#define HM    256
#define HMOD  64
#define MOD_IN  133
#define MOD_OUT 16512   // N*N + N

typedef __attribute__((ext_vector_type(16))) __bf16 v16bf;
typedef __attribute__((ext_vector_type(8)))  float  v8f;

union FragU { uint4 q[2]; v16bf v; };

__device__ __forceinline__ unsigned short f2bf(float f) {
    union { float f; unsigned u; } x; x.f = f;
    unsigned r = x.u + 0x7FFFu + ((x.u >> 16) & 1u);   // round-to-nearest-even
    return (unsigned short)(r >> 16);
}

// A fragment (16x32 bf16), source row-major [row][k] with row stride ldk (elements).
// Lane l: row = row0 + (l&15); K chunks {h*8..h*8+7} and {16+h*8..+7}, h = l>>4.
__device__ __forceinline__ v16bf load_a_frag(const unsigned short* base, int row0, int ldk, int k0) {
    int lane = threadIdx.x & 31;
    const unsigned short* p = base + (size_t)(row0 + (lane & 15)) * ldk + k0 + ((lane >> 4) << 3);
    FragU u;
    u.q[0] = *(const uint4*)(p);
    u.q[1] = *(const uint4*)(p + 16);
    return u.v;
}

// B fragment (32x16 bf16) from weights stored [n][k] row-major, stride ldk.
// Lane l: n = n0 + (l&15); K chunks {h*8..h*8+7} and {16+h*8..+7}, h = l>>4.
__device__ __forceinline__ v16bf load_b_frag(const unsigned short* base, int n0, int ldk, int k0) {
    int lane = threadIdx.x & 31;
    const unsigned short* p = base + (size_t)(n0 + (lane & 15)) * ldk + k0 + ((lane >> 4) << 3);
    FragU u;
    u.q[0] = *(const uint4*)(p);
    u.q[1] = *(const uint4*)(p + 16);
    return u.v;
}

// ---------------- K0: fp32 -> bf16 weight conversion ----------------
__global__ void k_cvt_bf16(const float* __restrict__ src, unsigned short* __restrict__ dst, int n) {
    int i = blockIdx.x * 256 + threadIdx.x;
    if (i < n) dst[i] = f2bf(src[i]);
}

// ---------------- Fused per-(b,c) kernel ----------------
// received = W@msg (+inject) kept in LDS (bf16) -> state MLP -> msg MLP,
// plus h_mean / msg_mean / mean|W| side-products for the modulator.
//
// LDS phase-aliased layout (bytes):
//   [0      , 49152 )  sIn   ushort[128*192]  state-MLP input (h | recv | nid)
//   [49152  , 53248 )  sInj  float[16*64]     zero-padded inject tile
//   [53248  , 53504 )  sColH float[64]        h_new column sums
//   [53504  , 53760 )  sColM float[64]        msg_new column sums
//   [53760  , 54784 )  sRed  float[256]       |W| reduction
//   [54784  , 136704)  region1:
//        phase A: sW ushort[128*128] @54784,  sMT ushort[64*128] @87552
//        phase B: sHid ushort[128*256] @54784, sHnew ushort[128*64] @120320
#define OFF_INJ   49152
#define OFF_COLH  53248
#define OFF_COLM  53504
#define OFF_RED   53760
#define OFF_R1    54784
#define SMEM_BYTES 136704

__global__ void __launch_bounds__(256) k_fused(
    const float* __restrict__ W, const float* __restrict__ msg,
    const float* __restrict__ h, const float* __restrict__ nid,
    const float* __restrict__ injw, const float* __restrict__ injb,
    const float* __restrict__ Haug,
    const unsigned short* __restrict__ sw1b, const float* __restrict__ sb1,
    const unsigned short* __restrict__ sw2b, const float* __restrict__ sb2,
    const unsigned short* __restrict__ mw1b, const float* __restrict__ mb1,
    const unsigned short* __restrict__ mw2b, const float* __restrict__ mb2,
    float* __restrict__ h_new, float* __restrict__ msg_new,
    float* __restrict__ hmean, float* __restrict__ mmean, float* __restrict__ wstat) {

    __shared__ __align__(16) char smem[SMEM_BYTES];
    unsigned short* sIn   = (unsigned short*)(smem);
    float*          sInj  = (float*)(smem + OFF_INJ);
    float*          sColH = (float*)(smem + OFF_COLH);
    float*          sColM = (float*)(smem + OFF_COLM);
    float*          sRed  = (float*)(smem + OFF_RED);
    unsigned short* sW    = (unsigned short*)(smem + OFF_R1);
    unsigned short* sMT   = (unsigned short*)(smem + OFF_R1 + 32768);
    unsigned short* sHid  = (unsigned short*)(smem + OFF_R1);            // aliases sW/sMT
    unsigned short* sHnew = (unsigned short*)(smem + OFF_R1 + 65536);

    int bc = blockIdx.x; int b = bc / NC, c = bc % NC;
    int tid = threadIdx.x;

    if (tid < 64) { sColH[tid] = 0.f; sColM[tid] = 0.f; }

    // ---- staging: W (with |W| partial), msg^T, h, nid, inject ----
    float wabs = 0.f;
    const float* Wp = W + (size_t)bc * (NN * NN);
    for (int i = tid; i < NN * NN; i += 256) {
        float v = Wp[i];
        wabs += fabsf(v);
        sW[i] = f2bf(v);
    }
    const float* Mp = msg + (size_t)bc * (NN * DN);
    for (int i = tid; i < NN * DN; i += 256) {
        int j = i >> 6, d = i & 63;
        sMT[d * NN + j] = f2bf(Mp[i]);
    }
    const float* hp = h   + (size_t)bc * (NN * DN);
    const float* np = nid + (size_t)c  * (NN * DN);
    for (int i = tid; i < NN * DN; i += 256) {
        int row = i >> 6, d = i & 63;
        sIn[row * 192 + d]       = f2bf(hp[i]);
        sIn[row * 192 + 128 + d] = f2bf(np[i]);
    }
    for (int i = tid; i < 16 * 64; i += 256) sInj[i] = 0.f;   // zero-pad tile
    if (tid < 128) {   // inject rows 0,1 (ALPHA*DN = 128 values)
        const float* iw = injw + ((size_t)c * 128 + tid) * DN;
        const float* cs = Haug + (size_t)b * NC * DN + (size_t)c * DN;
        float a = injb[c * 128 + tid];
        #pragma unroll 8
        for (int i = 0; i < DN; ++i) a += cs[i] * iw[i];
        sInj[tid] = a;
    }
    __syncthreads();   // B1: staging complete

    int wave = tid >> 5, lane = tid & 31, half = lane >> 4, ln = lane & 15;
    int m0 = wave * 16;
    int wave0 = (__builtin_amdgcn_readfirstlane(m0) == 0);   // scalar branch condition

    // ---- received = W @ msg, result written as bf16 into sIn cols 64..127 ----
    for (int tn = 0; tn < 4; ++tn) {
        v8f acc = {};
        #pragma unroll
        for (int kt = 0; kt < 4; ++kt) {
            v16bf af = load_a_frag(sW,  m0,      NN, kt * 32);
            v16bf bf = load_b_frag(sMT, tn * 16, NN, kt * 32);
            acc = __builtin_amdgcn_wmma_f32_16x16x32_bf16(false, af, false, bf,
                                                          (short)0, acc, false, false);
        }
        int n = tn * 16 + ln;
        if (wave0) {                   // SGPR condition: no EXEC manipulation
            #pragma unroll
            for (int r = 0; r < 8; ++r) acc[r] += sInj[(r + 8 * half) * 64 + n];
        }
        #pragma unroll
        for (int r = 0; r < 8; ++r) {
            int m = m0 + r + 8 * half;
            sIn[m * 192 + 64 + n] = f2bf(acc[r]);
        }
    }
    __syncthreads();   // B2: sW/sMT dead -> region1 may be re-used as sHid

    // ---- state MLP layer 1: [128x192] x [192->256], tanh -> sHid ----
    for (int tn = 0; tn < 16; ++tn) {
        v8f acc = {};
        #pragma unroll
        for (int kt = 0; kt < 6; ++kt) {
            v16bf af = load_a_frag(sIn, m0, 192, kt * 32);
            v16bf bf = load_b_frag(sw1b, tn * 16, 192, kt * 32);
            acc = __builtin_amdgcn_wmma_f32_16x16x32_bf16(false, af, false, bf,
                                                          (short)0, acc, false, false);
        }
        int n = tn * 16 + ln;
        float bias = sb1[n];
        #pragma unroll
        for (int r = 0; r < 8; ++r) {
            int m = m0 + r + 8 * half;
            sHid[m * HS + n] = f2bf(tanhf(acc[r] + bias));
        }
    }
    // (each wave reads only its own rows of sHid next -> no barrier needed)

    // ---- state MLP layer 2: [128x256] x [256->64], tanh -> h_new, sHnew, col sums ----
    float* outH = h_new + (size_t)bc * (NN * DN);
    for (int tn = 0; tn < 4; ++tn) {
        v8f acc = {};
        #pragma unroll
        for (int kt = 0; kt < 8; ++kt) {
            v16bf af = load_a_frag(sHid, m0, HS, kt * 32);
            v16bf bf = load_b_frag(sw2b, tn * 16, HS, kt * 32);
            acc = __builtin_amdgcn_wmma_f32_16x16x32_bf16(false, af, false, bf,
                                                          (short)0, acc, false, false);
        }
        int n = tn * 16 + ln;
        float bias = sb2[n];
        float csum = 0.f;
        #pragma unroll
        for (int r = 0; r < 8; ++r) {
            int m = m0 + r + 8 * half;
            float v = tanhf(acc[r] + bias);
            outH[m * DN + n] = v;
            sHnew[m * DN + n] = f2bf(v);
            csum += v;
        }
        atomicAdd(&sColH[n], csum);
    }
    // own-rows property again: msg layer1 reads only this wave's rows of sHnew/sIn

    // ---- msg MLP layer 1: [128x128] x [128->256], tanh -> sHid ----
    // K 0..63 from sHnew (h_new), K 64..127 from sIn recv columns.
    for (int tn = 0; tn < 16; ++tn) {
        v8f acc = {};
        {
            v16bf af = load_a_frag(sHnew, m0, DN, 0);
            v16bf bf = load_b_frag(mw1b, tn * 16, 128, 0);
            acc = __builtin_amdgcn_wmma_f32_16x16x32_bf16(false, af, false, bf,
                                                          (short)0, acc, false, false);
        }
        {
            v16bf af = load_a_frag(sHnew, m0, DN, 32);
            v16bf bf = load_b_frag(mw1b, tn * 16, 128, 32);
            acc = __builtin_amdgcn_wmma_f32_16x16x32_bf16(false, af, false, bf,
                                                          (short)0, acc, false, false);
        }
        {
            v16bf af = load_a_frag(sIn + 64, m0, 192, 0);
            v16bf bf = load_b_frag(mw1b, tn * 16, 128, 64);
            acc = __builtin_amdgcn_wmma_f32_16x16x32_bf16(false, af, false, bf,
                                                          (short)0, acc, false, false);
        }
        {
            v16bf af = load_a_frag(sIn + 64, m0, 192, 32);
            v16bf bf = load_b_frag(mw1b, tn * 16, 128, 96);
            acc = __builtin_amdgcn_wmma_f32_16x16x32_bf16(false, af, false, bf,
                                                          (short)0, acc, false, false);
        }
        int n = tn * 16 + ln;
        float bias = mb1[n];
        #pragma unroll
        for (int r = 0; r < 8; ++r) {
            int m = m0 + r + 8 * half;
            sHid[m * HM + n] = f2bf(tanhf(acc[r] + bias));
        }
    }

    // ---- msg MLP layer 2: [128x256] x [256->64], tanh -> msg_new, col sums ----
    float* outM = msg_new + (size_t)bc * (NN * DN);
    for (int tn = 0; tn < 4; ++tn) {
        v8f acc = {};
        #pragma unroll
        for (int kt = 0; kt < 8; ++kt) {
            v16bf af = load_a_frag(sHid, m0, HM, kt * 32);
            v16bf bf = load_b_frag(mw2b, tn * 16, HM, kt * 32);
            acc = __builtin_amdgcn_wmma_f32_16x16x32_bf16(false, af, false, bf,
                                                          (short)0, acc, false, false);
        }
        int n = tn * 16 + ln;
        float bias = mb2[n];
        float csum = 0.f;
        #pragma unroll
        for (int r = 0; r < 8; ++r) {
            int m = m0 + r + 8 * half;
            float v = tanhf(acc[r] + bias);
            outM[m * DN + n] = v;
            csum += v;
        }
        atomicAdd(&sColM[n], csum);
    }

    // ---- reductions: mean|W|, h_mean, msg_mean ----
    sRed[tid] = wabs;
    __syncthreads();   // also fences the ds_add_f32 column sums
    for (int off = 128; off > 0; off >>= 1) {
        if (tid < off) sRed[tid] += sRed[tid + off];
        __syncthreads();
    }
    if (tid == 0) wstat[bc] = sRed[0] * (1.0f / (NN * NN));
    if (tid < 64) {
        hmean[(size_t)bc * 64 + tid] = sColH[tid] * (1.0f / NN);
        mmean[(size_t)bc * 64 + tid] = sColM[tid] * (1.0f / NN);
    }
}

// ---------------- K4: modulator input + hidden (tiny) ----------------
__global__ void __launch_bounds__(64) k_mod_hidden(
    const float* __restrict__ hmean, const float* __restrict__ mmean,
    const float* __restrict__ wstat, const float* __restrict__ decay,
    const float* __restrict__ drift, const float* __restrict__ s1, const float* __restrict__ s2,
    const float* __restrict__ modw1, const float* __restrict__ modb1,
    float* __restrict__ hidden) {
    int bc = blockIdx.x; int b = bc / NC, c = bc % NC;
    __shared__ float sIn[MOD_IN];
    int tid = threadIdx.x;

    sIn[tid]      = hmean[(size_t)bc * 64 + tid];
    sIn[64 + tid] = mmean[(size_t)bc * 64 + tid];
    if (tid == 0) {
        sIn[128] = wstat[bc];
        const float* dp = decay + (size_t)bc * NN;
        float s = 0.f;
        for (int i = 0; i < NN; ++i) s += dp[i];
        sIn[129] = s * (1.0f / NN);
        sIn[130] = drift[bc];
        sIn[131] = s1[b];
        sIn[132] = s2[b];
    }
    __syncthreads();

    float acc = modb1[c * HMOD + tid];
    const float* w = modw1 + (size_t)c * MOD_IN * HMOD + tid;
    for (int i = 0; i < MOD_IN; ++i) acc += sIn[i] * w[i * HMOD];
    hidden[(size_t)bc * HMOD + tid] = tanhf(acc);
}

// ---------------- K5: mod_out expansion + apply to W / decay (HBM-bound) ----------------
__global__ void __launch_bounds__(256) k_mod_apply(
    const float* __restrict__ hidden, const float* __restrict__ modw2,
    const float* __restrict__ modb2, const float* __restrict__ W,
    const float* __restrict__ decay, float* __restrict__ outW, float* __restrict__ outDecay) {
    int c = blockIdx.y;
    int o = blockIdx.x * 256 + threadIdx.x;
    __shared__ float sH[BB * HMOD];
    for (int i = threadIdx.x; i < BB * HMOD; i += 256)
        sH[i] = hidden[(size_t)((i >> 6) * NC + c) * HMOD + (i & 63)];
    __syncthreads();
    if (o >= MOD_OUT) return;

    float acc[BB];
    #pragma unroll
    for (int b = 0; b < BB; ++b) acc[b] = 0.f;
    const float* w = modw2 + (size_t)c * HMOD * MOD_OUT + o;
    #pragma unroll 4
    for (int hh = 0; hh < HMOD - 4; ++hh) {       // main body with prefetch
        __builtin_prefetch(w + (size_t)(hh + 4) * MOD_OUT, 0, 1);
        float wv = w[(size_t)hh * MOD_OUT];
        #pragma unroll
        for (int b = 0; b < BB; ++b) acc[b] += sH[b * HMOD + hh] * wv;
    }
    #pragma unroll
    for (int hh = HMOD - 4; hh < HMOD; ++hh) {    // tail, no prefetch
        float wv = w[(size_t)hh * MOD_OUT];
        #pragma unroll
        for (int b = 0; b < BB; ++b) acc[b] += sH[b * HMOD + hh] * wv;
    }
    float bias = modb2[(size_t)c * MOD_OUT + o];
    if (o < NN * NN) {
        #pragma unroll
        for (int b = 0; b < BB; ++b) {
            size_t idx = (size_t)(b * NC + c) * (NN * NN) + o;
            outW[idx] = W[idx] + acc[b] + bias;
        }
    } else {
        int oo = o - NN * NN;
        #pragma unroll
        for (int b = 0; b < BB; ++b) {
            size_t idx = (size_t)(b * NC + c) * NN + oo;
            outDecay[idx] = decay[idx] + acc[b] + bias;
        }
    }
}

extern "C" void kernel_launch(void* const* d_in, const int* in_sizes, int n_in,
                              void* d_out, int out_size, void* d_ws, size_t ws_size,
                              hipStream_t stream) {
    const float* h     = (const float*)d_in[0];
    const float* msg   = (const float*)d_in[1];
    const float* W     = (const float*)d_in[2];
    const float* decay = (const float*)d_in[3];
    const float* drift = (const float*)d_in[4];
    const float* s1    = (const float*)d_in[5];
    const float* s2    = (const float*)d_in[6];
    const float* Haug  = (const float*)d_in[7];
    const float* nid   = (const float*)d_in[8];
    const float* sw1   = (const float*)d_in[9];
    const float* sb1   = (const float*)d_in[10];
    const float* sw2   = (const float*)d_in[11];
    const float* sb2   = (const float*)d_in[12];
    const float* mgw1  = (const float*)d_in[13];
    const float* mgb1  = (const float*)d_in[14];
    const float* mgw2  = (const float*)d_in[15];
    const float* mgb2  = (const float*)d_in[16];
    const float* injw  = (const float*)d_in[17];
    const float* injb  = (const float*)d_in[18];
    const float* modw1 = (const float*)d_in[19];
    const float* modb1 = (const float*)d_in[20];
    const float* modw2 = (const float*)d_in[21];
    const float* modb2 = (const float*)d_in[22];

    float* out       = (float*)d_out;
    float* h_new     = out;                        // B*NC*N*DN = 4194304
    float* msg_new   = out + 4194304;
    float* W_new     = out + 8388608;              // B*NC*N*N  = 8388608
    float* decay_new = out + 16777216;             // B*NC*N    = 65536

    char* ws = (char*)d_ws;
    float*          hidden = (float*)(ws);                      // 131072 B
    float*          hmean  = (float*)(ws + 131072);             // 131072 B
    float*          mmean  = (float*)(ws + 262144);             // 131072 B
    float*          wstat  = (float*)(ws + 393216);             // 2048 B
    unsigned short* sw1b   = (unsigned short*)(ws + 395264);    // 256*192 bf16
    unsigned short* sw2b   = (unsigned short*)(ws + 493568);    // 64*256  bf16
    unsigned short* mw1b   = (unsigned short*)(ws + 526336);    // 256*128 bf16
    unsigned short* mw2b   = (unsigned short*)(ws + 591872);    // 64*256  bf16

    k_cvt_bf16<<<(HS * 192 + 255) / 256, 256, 0, stream>>>(sw1, sw1b, HS * 192);
    k_cvt_bf16<<<(DN * HS + 255) / 256, 256, 0, stream>>>(sw2, sw2b, DN * HS);
    k_cvt_bf16<<<(HM * 128 + 255) / 256, 256, 0, stream>>>(mgw1, mw1b, HM * 128);
    k_cvt_bf16<<<(DN * HM + 255) / 256, 256, 0, stream>>>(mgw2, mw2b, DN * HM);

    k_fused<<<BB * NC, 256, 0, stream>>>(W, msg, h, nid, injw, injb, Haug,
                                         sw1b, sb1, sw2b, sb2,
                                         mw1b, mgb1, mw2b, mgb2,
                                         h_new, msg_new, hmean, mmean, wstat);

    k_mod_hidden<<<BB * NC, 64, 0, stream>>>(hmean, mmean, wstat, decay, drift, s1, s2,
                                             modw1, modb1, hidden);

    k_mod_apply<<<dim3((MOD_OUT + 255) / 256, NC), 256, 0, stream>>>(
        hidden, modw2, modb2, W, decay, W_new, decay_new);
}
